// Decoder_11364483465840
// MI455X (gfx1250) — matrix operand order repaired and checked by reading, not compiled, not run on hardware
//
#include <hip/hip_runtime.h>
#include <hip/hip_bf16.h>
#include <stdint.h>

// Problem constants (fixed by the reference)
#define T_ENC 400
#define NB    64
#define LDEC  250
#define HID   512
#define KSZ   512
#define VSZ   512
#define VOC   1024
#define K1    1536   // [e | context | h1] width for LSTM1
#define K2    1024   // [h1 | h2] width for LSTM2
#define K3    1024   // [h2 | context] width for out-proj
#define MG    2048   // 4*HID gate width

typedef __attribute__((ext_vector_type(16))) __bf16 v16bf;
typedef __attribute__((ext_vector_type(8)))  float  v8f;

union FragBF {
  v16bf v;
  uint4 u[2];
};

__device__ __forceinline__ float sigmoidf_(float x) { return 1.0f / (1.0f + __expf(-x)); }

// ---------------------------------------------------------------------------
// Prepack kernels (run once per launch, weights -> bf16, concatenated in K)
// ---------------------------------------------------------------------------
__global__ void pack2_kernel(const float* __restrict__ A, int Ka,
                             const float* __restrict__ B, int Kb,
                             __hip_bfloat16* __restrict__ dst, int M) {
  const int K = Ka + Kb;
  const size_t total = (size_t)M * K;
  for (size_t i = (size_t)blockIdx.x * blockDim.x + threadIdx.x; i < total;
       i += (size_t)gridDim.x * blockDim.x) {
    int m = (int)(i / K);
    int k = (int)(i % K);
    float v = (k < Ka) ? A[(size_t)m * Ka + k] : B[(size_t)m * Kb + (k - Ka)];
    dst[i] = __float2bfloat16(v);
  }
}

__global__ void cvt_kernel(const float* __restrict__ src,
                           __hip_bfloat16* __restrict__ dst, size_t n) {
  for (size_t i = (size_t)blockIdx.x * blockDim.x + threadIdx.x; i < n;
       i += (size_t)gridDim.x * blockDim.x)
    dst[i] = __float2bfloat16(src[i]);
}

__global__ void addb_kernel(const float* __restrict__ a, const float* __restrict__ b,
                            float* __restrict__ dst, int n) {
  int i = blockIdx.x * blockDim.x + threadIdx.x;
  if (i < n) dst[i] = a[i] + b[i];
}

__global__ void zero_kernel(uint32_t* __restrict__ p, size_t nwords) {
  for (size_t i = (size_t)blockIdx.x * blockDim.x + threadIdx.x; i < nwords;
       i += (size_t)gridDim.x * blockDim.x)
    p[i] = 0u;
}

// ---------------------------------------------------------------------------
// Attention + embedding + state-forwarding kernel. One block per batch sample.
//   - copies prev-step h1/h2 bf16 into the GEMM input buffers
//   - e = embedding[text[n,t]]; energy over T; masked softmax; context
//   - writes bf16 e/context into Xcat1, context into Xcat3
// ---------------------------------------------------------------------------
__global__ __launch_bounds__(512)
void attn_kernel(const float* __restrict__ key_proj,
                 const float* __restrict__ values,
                 const int* __restrict__ text,
                 const int* __restrict__ text_lens,
                 const float* __restrict__ embedding,
                 __hip_bfloat16* __restrict__ Xc1,
                 __hip_bfloat16* __restrict__ Xc2,
                 __hip_bfloat16* __restrict__ Xc3,
                 int t) {
  const int n   = blockIdx.x;
  const int tid = threadIdx.x;
  const int wave = tid >> 5;
  const int lane = tid & 31;

  __shared__ float e_s[HID];
  __shared__ float energy_s[T_ENC];
  __shared__ float red_s[512];

  // embedding gather + forward previous hidden states (bf16 copies)
  const int tok = text[n * LDEC + t];
  const float ev_in = embedding[(size_t)tok * HID + tid];
  e_s[tid] = ev_in;
  Xc1[(size_t)n * K1 + tid]        = __float2bfloat16(ev_in);     // e slice
  Xc1[(size_t)n * K1 + 1024 + tid] = Xc2[(size_t)n * K2 + tid];   // h1_prev
  Xc2[(size_t)n * K2 + 512 + tid]  = Xc3[(size_t)n * K3 + tid];   // h2_prev
  __syncthreads();

  // energies: one encoder step per wave-iteration, lanes across KSZ
  for (int ti = wave; ti < T_ENC; ti += 16) {
    const float* kp = key_proj + ((size_t)ti * NB + n) * KSZ;
    float p = 0.0f;
    #pragma unroll 4
    for (int k = lane; k < KSZ; k += 32) p += kp[k] * e_s[k];
    #pragma unroll
    for (int m = 16; m >= 1; m >>= 1) p += __shfl_xor(p, m, 32);
    if (lane == 0) energy_s[ti] = p;
  }
  __syncthreads();

  // masked softmax over T (== softmax * mask / sum in the reference)
  const int len = text_lens[n];
  float mv = -1e30f;
  if (tid < T_ENC && tid < len) mv = energy_s[tid];
  red_s[tid] = mv;
  __syncthreads();
  for (int s = 256; s > 0; s >>= 1) {
    if (tid < s) red_s[tid] = fmaxf(red_s[tid], red_s[tid + s]);
    __syncthreads();
  }
  const float maxv = red_s[0];
  __syncthreads();
  float ex = 0.0f;
  if (tid < T_ENC && tid < len) ex = __expf(energy_s[tid] - maxv);
  if (tid < T_ENC) energy_s[tid] = ex;
  red_s[tid] = ex;
  __syncthreads();
  for (int s = 256; s > 0; s >>= 1) {
    if (tid < s) red_s[tid] += red_s[tid + s];
    __syncthreads();
  }
  const float inv = 1.0f / red_s[0];
  if (tid < T_ENC) energy_s[tid] *= inv;
  __syncthreads();

  // context: each thread owns one v-channel, coalesced over values
  float acc = 0.0f;
  for (int ti = 0; ti < len; ++ti)
    acc += energy_s[ti] * values[((size_t)ti * NB + n) * VSZ + tid];
  const __hip_bfloat16 cb = __float2bfloat16(acc);
  Xc1[(size_t)n * K1 + 512 + tid] = cb;   // context into LSTM1 input
  Xc3[(size_t)n * K3 + 512 + tid] = cb;   // context into out-proj input
}

// ---------------------------------------------------------------------------
// Fused LSTM cell: gates = Xin(64xK) @ W(2048xK)^T + bias, then pointwise.
// Block = 4 waves; wave g computes the 16x16 gate-g tile for hidden chunk j.
// Uses v_wmma_f32_16x16x32_bf16. Fragment layouts per CDNA5 ISA.
// ---------------------------------------------------------------------------
__global__ __launch_bounds__(128)
void lstm_kernel(const __hip_bfloat16* __restrict__ Xin, int K,
                 const __hip_bfloat16* __restrict__ W,
                 const float* __restrict__ bias,
                 float* __restrict__ c,
                 __hip_bfloat16* __restrict__ dst0, int stride0,
                 __hip_bfloat16* __restrict__ dst1, int stride1) {
  const int tid  = threadIdx.x;
  const int wave = tid >> 5;          // gate index 0..3 (i,f,g,o)
  const int lane = tid & 31;
  const int r    = blockIdx.x & 3;    // batch row tile (16 rows)
  const int j    = blockIdx.x >> 2;   // hidden chunk 0..31
  const int n0   = r * 16;
  const int m0   = wave * HID + j * 16;

  const int rowA = n0 + (lane & 15);
  const int rowB = m0 + (lane & 15);
  const int loA  = (lane < 16) ? 0 : 8;    // A low-K base (hi base = loA+16)
  const int kbB  = (lane < 16) ? 0 : 8;    // B K base

  const __hip_bfloat16* pa = Xin + (size_t)rowA * K;
  const __hip_bfloat16* pb = W   + (size_t)rowB * K;

  v8f acc = {0.f, 0.f, 0.f, 0.f, 0.f, 0.f, 0.f, 0.f};
  for (int k0 = 0; k0 < K; k0 += 32) {
    FragBF a, b;
    a.u[0] = *reinterpret_cast<const uint4*>(pa + k0 + loA);
    a.u[1] = *reinterpret_cast<const uint4*>(pa + k0 + loA + 16);
    b.u[0] = *reinterpret_cast<const uint4*>(pb + k0 + kbB);
    b.u[1] = *reinterpret_cast<const uint4*>(pb + k0 + kbB + 16);
    __builtin_prefetch((const void*)(pb + k0 + 32 + kbB), 0, 3); // global_prefetch_b8
    acc = __builtin_amdgcn_wmma_f32_16x16x32_bf16(
        false, a.v, false, b.v, (short)0, acc, false, false);
  }

  // stash 4 gate tiles (f32) in LDS, then fused cell update
  __shared__ float g_s[4][16][17];
  const float bs = bias[m0 + (lane & 15)];
  const int nrow = (lane >> 4) << 3;   // 0 or 8
  #pragma unroll
  for (int i = 0; i < 8; ++i)
    g_s[wave][nrow + i][lane & 15] = acc[i] + bs;
  __syncthreads();

  for (int e = tid; e < 256; e += 128) {
    const int nl = e >> 4, kl = e & 15;
    const float ig = g_s[0][nl][kl];
    const float fg = g_s[1][nl][kl];
    const float gg = g_s[2][nl][kl];
    const float og = g_s[3][nl][kl];
    const int n  = n0 + nl;
    const int kh = j * 16 + kl;
    const float cp = c[n * HID + kh];
    const float cn = sigmoidf_(fg) * cp + sigmoidf_(ig) * tanhf(gg);
    const float h  = sigmoidf_(og) * tanhf(cn);
    c[n * HID + kh] = cn;
    const __hip_bfloat16 hb = __float2bfloat16(h);
    dst0[(size_t)n * stride0 + kh] = hb;
    if (dst1) dst1[(size_t)n * stride1 + kh] = hb;
  }
}

// ---------------------------------------------------------------------------
// Output projection: pred(64x1024) = Xc3(64x1024) @ Wout(1024x1024)^T + b_out
// ---------------------------------------------------------------------------
__global__ __launch_bounds__(128)
void outproj_kernel(const __hip_bfloat16* __restrict__ Xin,
                    const __hip_bfloat16* __restrict__ W,
                    const float* __restrict__ b_out,
                    float* __restrict__ out, int t) {
  const int tid  = threadIdx.x;
  const int wave = tid >> 5;
  const int lane = tid & 31;
  const int r    = blockIdx.x & 3;                    // batch row tile
  const int ct   = (blockIdx.x >> 2) * 4 + wave;      // vocab col tile 0..63
  const int n0   = r * 16;
  const int col0 = ct * 16;

  const int rowA = n0 + (lane & 15);
  const int rowB = col0 + (lane & 15);
  const int loA  = (lane < 16) ? 0 : 8;
  const int kbB  = (lane < 16) ? 0 : 8;

  const __hip_bfloat16* pa = Xin + (size_t)rowA * K3;
  const __hip_bfloat16* pb = W   + (size_t)rowB * K3;

  v8f acc = {0.f, 0.f, 0.f, 0.f, 0.f, 0.f, 0.f, 0.f};
  for (int k0 = 0; k0 < K3; k0 += 32) {
    FragBF a, b;
    a.u[0] = *reinterpret_cast<const uint4*>(pa + k0 + loA);
    a.u[1] = *reinterpret_cast<const uint4*>(pa + k0 + loA + 16);
    b.u[0] = *reinterpret_cast<const uint4*>(pb + k0 + kbB);
    b.u[1] = *reinterpret_cast<const uint4*>(pb + k0 + kbB + 16);
    __builtin_prefetch((const void*)(pb + k0 + 32 + kbB), 0, 3);
    acc = __builtin_amdgcn_wmma_f32_16x16x32_bf16(
        false, a.v, false, b.v, (short)0, acc, false, false);
  }

  const int colL = lane & 15;
  const float bo = b_out[col0 + colL];
  #pragma unroll
  for (int i = 0; i < 8; ++i) {
    const int n = n0 + ((lane >> 4) << 3) + i;
    out[(size_t)n * LDEC * VOC + (size_t)t * VOC + col0 + colL] = acc[i] + bo;
  }
}

// ---------------------------------------------------------------------------
// Host launch
// ---------------------------------------------------------------------------
extern "C" void kernel_launch(void* const* d_in, const int* in_sizes, int n_in,
                              void* d_out, int out_size, void* d_ws, size_t ws_size,
                              hipStream_t stream) {
  (void)in_sizes; (void)n_in; (void)out_size; (void)ws_size;

  const float* key_proj  = (const float*)d_in[0];
  const float* values    = (const float*)d_in[1];
  const int*   text      = (const int*)d_in[2];
  const int*   text_lens = (const int*)d_in[3];
  const float* embedding = (const float*)d_in[4];
  const float* W_ih1     = (const float*)d_in[5];
  const float* W_hh1     = (const float*)d_in[6];
  const float* b_ih1     = (const float*)d_in[7];
  const float* b_hh1     = (const float*)d_in[8];
  const float* W_ih2     = (const float*)d_in[9];
  const float* W_hh2     = (const float*)d_in[10];
  const float* b_ih2     = (const float*)d_in[11];
  const float* b_out     = (const float*)d_in[14];
  const float* W_out     = (const float*)d_in[13];
  const float* b_hh2     = (const float*)d_in[12];
  float* out = (float*)d_out;

  // Workspace layout (bytes), all 16B aligned. ~12.7 MB total.
  char* ws = (char*)d_ws;
  __hip_bfloat16* Wcat1 = (__hip_bfloat16*)(ws + 0);              // 2048x1536 bf16
  __hip_bfloat16* Wcat2 = (__hip_bfloat16*)(ws + 6291456);        // 2048x1024 bf16
  __hip_bfloat16* WoutB = (__hip_bfloat16*)(ws + 10485760);       // 1024x1024 bf16
  float*          bcat1 = (float*)(ws + 12582912);                // 2048 f32
  float*          bcat2 = (float*)(ws + 12591104);                // 2048 f32
  float*          c1    = (float*)(ws + 12599296);                // 64x512 f32
  float*          c2    = (float*)(ws + 12730368);                // 64x512 f32
  __hip_bfloat16* Xc1   = (__hip_bfloat16*)(ws + 12861440);       // 64x1536 bf16
  __hip_bfloat16* Xc2   = (__hip_bfloat16*)(ws + 13058048);       // 64x1024 bf16
  __hip_bfloat16* Xc3   = (__hip_bfloat16*)(ws + 13189120);       // 64x1024 bf16

  // --- one-time prepack (re-run every call: idempotent, reads inputs only) ---
  pack2_kernel<<<512, 256, 0, stream>>>(W_ih1, 1024, W_hh1, 512, Wcat1, MG);
  pack2_kernel<<<512, 256, 0, stream>>>(W_ih2, 512,  W_hh2, 512, Wcat2, MG);
  cvt_kernel<<<512, 256, 0, stream>>>(W_out, WoutB, (size_t)VOC * K3);
  addb_kernel<<<8, 256, 0, stream>>>(b_ih1, b_hh1, bcat1, MG);
  addb_kernel<<<8, 256, 0, stream>>>(b_ih2, b_hh2, bcat2, MG);
  // zero c1,c2 + all Xcat buffers (contiguous region)
  zero_kernel<<<256, 256, 0, stream>>>((uint32_t*)c1, (size_t)720896 / 4);

  // --- sequential decode: 4 stream-ordered kernels per step ---
  for (int t = 0; t < LDEC; ++t) {
    attn_kernel<<<NB, 512, 0, stream>>>(key_proj, values, text, text_lens,
                                        embedding, Xc1, Xc2, Xc3, t);
    lstm_kernel<<<128, 128, 0, stream>>>(Xc1, K1, Wcat1, bcat1, c1,
                                         Xc1 + 1024, K1,   // h1 -> Xcat1[:,1024:]
                                         Xc2, K2);          // h1 -> Xcat2[:,0:512]
    lstm_kernel<<<128, 128, 0, stream>>>(Xc2, K2, Wcat2, bcat2, c2,
                                         Xc3, K3,           // h2 -> Xcat3[:,0:512]
                                         nullptr, 0);
    outproj_kernel<<<64, 128, 0, stream>>>(Xc3, WoutB, b_out, out, t);
  }
}
